// Primal_attention_10582799417403
// MI455X (gfx1250) — matrix operand order repaired and checked by reading, not compiled, not run on hardware
//
#include <hip/hip_runtime.h>
#include <hip/hip_bf16.h>
#include <stdint.h>

// Problem constants (match reference)
constexpr int Bc  = 4;
constexpr int Lc  = 4096;
constexpr int Dc  = 1024;
constexpr int Hc  = 16;
constexpr int Rc  = 30;
constexpr int DKc = 64;
constexpr int BLc  = Bc * Lc;              // 16384 GEMM rows
constexpr int BLD  = Bc * Lc * Dc;         // 16,777,216
constexpr int BHLR = Bc * Hc * Lc * Rc;    // 7,864,320

// chunked-scan parameters for the causal cumulative mean
constexpr int NCH = 16;                    // chunks along L
constexpr int CL  = Lc / NCH;              // 256 elems per chunk

typedef __attribute__((ext_vector_type(16))) __bf16 v16bf;
typedef __attribute__((ext_vector_type(8)))  __bf16 v8bf;
typedef __attribute__((ext_vector_type(8)))  float  v8f;

// k-permutation inside each 32-element group: swap bits 3 and 4.
// Makes each lane's 16-element WMMA A-fragment contiguous (ISA 7.12.2).
__device__ __forceinline__ size_t swap34(size_t i) {
    return (i & ~(size_t)24) | ((i & 8) << 1) | ((i & 16) >> 1);
}

__device__ __forceinline__ unsigned lds_lo32(const void* p) {
    // generic LDS pointer = {shared aperture, lds offset}; low 32 bits = LDS addr
    return (unsigned)(unsigned long long)p;
}
__device__ __forceinline__ void async_copy_b128(unsigned lds_addr, const void* g) {
    asm volatile("global_load_async_to_lds_b128 %0, %1, off"
                 :: "v"(lds_addr), "v"((unsigned long long)g) : "memory");
}
__device__ __forceinline__ void wait_async0() {
    asm volatile("s_wait_asynccnt 0" ::: "memory");
}

// ---------------------------------------------------------------------------
// GEMM: C[M,1024] = A[M,1024](bf16, k-permuted) @ B(bf16, stored [N,K]) + bias
// BM=128, BN=64, BK=32; 256 threads = 8 waves; wave = (row-half, 16-col strip).
// A tiles: double-buffered LDS via global_load_async_to_lds_b128 (ASYNCcnt).
// B fragments: two direct 16B v8bf loads from global (weights live in L2).
// ---------------------------------------------------------------------------
#define GBM 128
#define GBN 64
#define GBK 32
constexpr int APITCH = 40;                 // bf16/row: 32 data + 8 pad = 80 B

__global__ __launch_bounds__(256)
void gemm_bf16_wmma(const __bf16* __restrict__ A,   // [M,K], k-permuted
                    const __bf16* __restrict__ BT,  // [N,K], plain
                    const float*  __restrict__ bias,
                    float* __restrict__ C) {
    constexpr int K = Dc, N = Dc;
    __shared__ __bf16 As[2][GBM * APITCH];

    const int tid  = threadIdx.x;
    const int wave = tid >> 5, lane = tid & 31;
    const int m0 = blockIdx.y * GBM, n0 = blockIdx.x * GBN;
    const int wc = wave & 3;               // column strip (16 cols)
    const int wr = wave >> 2;              // row half (64 rows)
    const int nlo = lane & 15, khl = lane >> 4;

    v8f acc[4] = {};

    // 512 16-byte chunks per tile, 2 per thread
    const int cr = tid >> 2, ccol = (tid & 3) * 8;   // rows 0..63 / elem col

    auto issue = [&](int buf, int kb) {
#pragma unroll
        for (int s = 0; s < 2; ++s) {
            int r = cr + s * 64;
            const __bf16* g = A + (size_t)(m0 + r) * K + kb + ccol;
            unsigned l = lds_lo32(&As[buf][r * APITCH + ccol]);
            async_copy_b128(l, g);
        }
    };

    issue(0, 0);
    int buf = 0;
    for (int kb = 0; kb < K; kb += GBK) {
        wait_async0();
        __syncthreads();
        if (kb + GBK < K) issue(buf ^ 1, kb + GBK);

        // B fragment: k = khl*8 .. +7 and 16 + khl*8 .. +7 of column n
        const __bf16* bp = BT + (size_t)(n0 + wc * 16 + nlo) * K + kb + khl * 8;
        const v8bf blo = *reinterpret_cast<const v8bf*>(bp);
        const v8bf bhi = *reinterpret_cast<const v8bf*>(bp + 16);
        const v16bf bfrag = __builtin_shufflevector(blo, bhi, 0, 1, 2, 3, 4, 5, 6, 7, 8, 9, 10, 11, 12, 13, 14, 15);

        // preload all 4 A fragments (DS loads issue as a clause, batched wait)
        v16bf af[4];
#pragma unroll
        for (int t = 0; t < 4; ++t) {
            const int mrow = wr * 64 + t * 16 + nlo;
            af[t] = *reinterpret_cast<const v16bf*>(
                &As[buf][mrow * APITCH + khl * 16]);
        }
#pragma unroll
        for (int t = 0; t < 4; ++t)
            acc[t] = __builtin_amdgcn_wmma_f32_16x16x32_bf16(
                false, af[t], false, bfrag, (short)0, acc[t], false, false);
        buf ^= 1;
    }

    // C/D layout: VGPR r -> row = (lane>>4)*8 + r, col = lane&15
    const float bsv = bias[n0 + wc * 16 + nlo];
#pragma unroll
    for (int t = 0; t < 4; ++t)
#pragma unroll
        for (int r = 0; r < 8; ++r) {
            int row = m0 + wr * 64 + t * 16 + khl * 8 + r;
            C[(size_t)row * N + n0 + wc * 16 + nlo] = acc[t][r] + bsv;
        }
}

// ---------------------------------------------------------------------------
// fp32 -> bf16 with k-permutation (bits 3<->4). Row length (1024) is a
// multiple of 32, so permuting the flat index is safe. 8 elems/thread.
// ---------------------------------------------------------------------------
__global__ __launch_bounds__(256)
void cvt_perm_kernel(const float* __restrict__ in, __bf16* __restrict__ out) {
    size_t i8 = ((size_t)blockIdx.x * 256 + threadIdx.x) * 8;
    const float4 a = *reinterpret_cast<const float4*>(in + i8);
    const float4 b = *reinterpret_cast<const float4*>(in + i8 + 4);
    v8bf v;
    v[0] = (__bf16)a.x; v[1] = (__bf16)a.y; v[2] = (__bf16)a.z; v[3] = (__bf16)a.w;
    v[4] = (__bf16)b.x; v[5] = (__bf16)b.y; v[6] = (__bf16)b.z; v[7] = (__bf16)b.w;
    *reinterpret_cast<v8bf*>(out + swap34(i8)) = v;
}

// ---------------------------------------------------------------------------
// Weight prep: W[K,N] fp32 -> WT[N,K] bf16 via 64x64 LDS tile transpose.
// ---------------------------------------------------------------------------
__global__ __launch_bounds__(256)
void wt_transpose_kernel(const float* __restrict__ W, __bf16* __restrict__ WT) {
    __shared__ __bf16 t[64][64 + 8];
    const int tid = threadIdx.x;
    const int k0 = blockIdx.y * 64, n0 = blockIdx.x * 64;
    for (int i = tid; i < 64 * 64; i += 256) {
        int r = i >> 6, c = i & 63;
        t[r][c] = (__bf16)W[(size_t)(k0 + r) * Dc + n0 + c];
    }
    __syncthreads();
    for (int i = tid; i < 64 * 64; i += 256) {
        int r = i >> 6, c = i & 63;
        WT[(size_t)(n0 + r) * Dc + k0 + c] = t[c][r];
    }
}

// ---------------------------------------------------------------------------
// Causal cumulative mean, 3-phase chunked scan over L (16x the parallelism
// of a serial per-channel scan). Thread layout (t, b, c, d), d fastest.
// ---------------------------------------------------------------------------
__global__ __launch_bounds__(256)
void scan_phase1(const float* __restrict__ qbuf, const float* __restrict__ kbuf,
                 float* __restrict__ csum) {
    int idx = blockIdx.x * 256 + threadIdx.x;      // 2*B*NCH*D threads
    int d = idx % Dc;
    int c = (idx / Dc) % NCH;
    int b = (idx / (Dc * NCH)) % Bc;
    int t = idx / (Dc * NCH * Bc);
    const float* buf = t ? kbuf : qbuf;
    size_t base = ((size_t)b * Lc + (size_t)c * CL) * Dc + d;
    float s = 0.f;
    for (int l = 0; l < CL; ++l) s += buf[base + (size_t)l * Dc];
    csum[idx] = s;
}

__global__ __launch_bounds__(256)
void scan_phase2(float* __restrict__ csum) {
    int idx = blockIdx.x * 256 + threadIdx.x;      // 2*B*D threads
    int d = idx % Dc;
    int b = (idx / Dc) % Bc;
    int t = idx / (Dc * Bc);
    size_t base = (((size_t)t * Bc + b) * NCH) * Dc + d;
    float run = 0.f;
#pragma unroll
    for (int c = 0; c < NCH; ++c) {
        float v = csum[base + (size_t)c * Dc];
        csum[base + (size_t)c * Dc] = run;         // exclusive prefix
        run += v;
    }
}

__global__ __launch_bounds__(256)
void scan_phase3(float* __restrict__ qbuf, float* __restrict__ kbuf,
                 const float* __restrict__ csum) {
    int idx = blockIdx.x * 256 + threadIdx.x;      // 2*B*NCH*D threads
    int d = idx % Dc;
    int c = (idx / Dc) % NCH;
    int b = (idx / (Dc * NCH)) % Bc;
    int t = idx / (Dc * NCH * Bc);
    float* buf = t ? kbuf : qbuf;
    float run = csum[idx];
    size_t base = ((size_t)b * Lc + (size_t)c * CL) * Dc + d;
    int l0 = c * CL;
    for (int l = 0; l < CL; ++l) {
        run += buf[base + (size_t)l * Dc];
        buf[base + (size_t)l * Dc] = run / (float)(l0 + l + 1);
    }
}

// ---------------------------------------------------------------------------
// L2-normalize over DK=64, in place; one wave per (b,l,h).
// ---------------------------------------------------------------------------
__global__ __launch_bounds__(256)
void l2norm_kernel(float* __restrict__ qbuf, float* __restrict__ kbuf) {
    float* buf = blockIdx.y ? kbuf : qbuf;
    int gw   = blockIdx.x * 8 + (threadIdx.x >> 5);
    int lane = threadIdx.x & 31;
    int h = gw % Hc;
    int l = (gw / Hc) % Lc;
    int b = gw / (Hc * Lc);
    size_t base = ((size_t)b * Lc + l) * Dc + h * DKc;
    float v0 = buf[base + lane];
    float v1 = buf[base + lane + 32];
    float s = v0 * v0 + v1 * v1;
#pragma unroll
    for (int m = 16; m; m >>= 1) s += __shfl_xor(s, m, 32);
    float inv = 1.0f / fmaxf(sqrtf(s), 1e-12f);
    buf[base + lane]      = v0 * inv;
    buf[base + lane + 32] = v1 * inv;
}

// ---------------------------------------------------------------------------
// Low-rank head projection: out[b,h,l,r] = sum_dk X[b,l,h*64+dk] * W[h,dk,r]
// ---------------------------------------------------------------------------
__global__ __launch_bounds__(256)
void lowrank_kernel(const float* __restrict__ X, const float* __restrict__ W,
                    float* __restrict__ out) {
    __shared__ float Ws[DKc * Rc];
    __shared__ float Xs[64][DKc + 1];
    const int tid = threadIdx.x;
    const int b = blockIdx.z, h = blockIdx.y, l0 = blockIdx.x * 64;

    for (int i = tid; i < DKc * Rc; i += 256)
        Ws[i] = W[(size_t)h * DKc * Rc + i];
    for (int i = tid; i < 64 * DKc; i += 256) {
        int row = i >> 6, col = i & 63;
        Xs[row][col] = X[((size_t)b * Lc + l0 + row) * Dc + h * DKc + col];
    }
    __syncthreads();

    for (int i = tid; i < 64 * Rc; i += 256) {
        int ll = i / Rc, r = i % Rc;
        float s = 0.f;
#pragma unroll 8
        for (int d = 0; d < DKc; ++d) s += Xs[ll][d] * Ws[d * Rc + r];
        out[(((size_t)b * Hc + h) * Lc + l0 + ll) * Rc + r] = s;
    }
}

// ---------------------------------------------------------------------------
// score @ Wc + bc -> bf16 activation in [B,L,D] layout, k-permuted, feeding
// the final WMMA GEMM directly.
// ---------------------------------------------------------------------------
__global__ __launch_bounds__(256)
void mix_kernel(const float* __restrict__ esc, const float* __restrict__ rsc,
                const float* __restrict__ Wc, const float* __restrict__ bc,
                __bf16* __restrict__ outh) {
    __shared__ float Wcs[2 * Rc * DKc];    // 60 x 64
    __shared__ float bcs[DKc];
    const int tid = threadIdx.x;
    const int b = blockIdx.y, l0 = blockIdx.x * 64;

    for (int i = tid; i < 2 * Rc * DKc; i += 256) Wcs[i] = Wc[i];
    if (tid < DKc) bcs[tid] = bc[tid];
    __syncthreads();

    for (int i = tid; i < 64 * Dc; i += 256) {
        int ll = i >> 10, d = i & (Dc - 1);
        int h = d >> 6, dk = d & 63;
        size_t srow = (((size_t)b * Hc + h) * Lc + l0 + ll) * Rc;
        float s = bcs[dk];
#pragma unroll 6
        for (int r = 0; r < Rc; ++r)
            s += esc[srow + r] * Wcs[r * DKc + dk]
               + rsc[srow + r] * Wcs[(Rc + r) * DKc + dk];
        size_t idx = ((size_t)b * Lc + l0 + ll) * Dc + d;
        outh[swap34(idx)] = (__bf16)s;
    }
}

// ---------------------------------------------------------------------------
extern "C" void kernel_launch(void* const* d_in, const int* in_sizes, int n_in,
                              void* d_out, int out_size, void* d_ws, size_t ws_size,
                              hipStream_t stream) {
    (void)in_sizes; (void)n_in; (void)out_size; (void)ws_size;

    const float* queries = (const float*)d_in[0];
    const float* keys    = (const float*)d_in[1];
    // d_in[2] = values (unused by the reference output)
    const float* Wq  = (const float*)d_in[3];
    const float* bq  = (const float*)d_in[4];
    const float* Wk  = (const float*)d_in[5];
    const float* bk  = (const float*)d_in[6];
    const float* We  = (const float*)d_in[7];
    const float* Wr  = (const float*)d_in[8];
    const float* Lam = (const float*)d_in[9];
    const float* Wc  = (const float*)d_in[10];
    const float* bc  = (const float*)d_in[11];
    const float* Wo  = (const float*)d_in[12];
    const float* bo  = (const float*)d_in[13];

    // workspace
    float*  qbuf = (float*)d_ws;            // BLD fp32
    float*  kbuf = qbuf + BLD;              // BLD fp32
    __bf16* qh   = (__bf16*)(kbuf + BLD);   // BLD bf16 (k-permuted)
    __bf16* kh   = qh + BLD;
    __bf16* hh   = kh + BLD;
    __bf16* WqT  = hh + BLD;                // Dc*Dc bf16 each
    __bf16* WkT  = WqT + Dc * Dc;
    __bf16* WoT  = WkT + Dc * Dc;
    float*  csum = (float*)(WoT + Dc * Dc); // 2*B*NCH*D fp32

    // output partitions: (out, escore, rscore, Lam)
    float* out = (float*)d_out;
    float* esc = out + BLD;
    float* rsc = esc + BHLR;
    float* lam = rsc + BHLR;

    dim3 blk(256);
    dim3 ggrid(Dc / GBN, BLc / GBM);        // (16, 128)
    dim3 wgrid(Dc / 64, Dc / 64);           // (16, 16)
    const int scanThreads = 2 * Bc * NCH * Dc;   // 131072

    // 0) bf16 prep: activations (k-permuted) and transposed weights
    cvt_perm_kernel<<<dim3(BLD / (8 * 256)), blk, 0, stream>>>(queries, qh);
    cvt_perm_kernel<<<dim3(BLD / (8 * 256)), blk, 0, stream>>>(keys, kh);
    wt_transpose_kernel<<<wgrid, blk, 0, stream>>>(Wq, WqT);
    wt_transpose_kernel<<<wgrid, blk, 0, stream>>>(Wk, WkT);
    wt_transpose_kernel<<<wgrid, blk, 0, stream>>>(Wo, WoT);

    // 1-2) projections (WMMA bf16, async-to-LDS pipelined)
    gemm_bf16_wmma<<<ggrid, blk, 0, stream>>>(qh, WqT, bq, qbuf);
    gemm_bf16_wmma<<<ggrid, blk, 0, stream>>>(kh, WkT, bk, kbuf);

    // 3) causal cumulative mean: 3-phase chunked scan (both tensors)
    scan_phase1<<<dim3(scanThreads / 256), blk, 0, stream>>>(qbuf, kbuf, csum);
    scan_phase2<<<dim3((2 * Bc * Dc) / 256), blk, 0, stream>>>(csum);
    scan_phase3<<<dim3(scanThreads / 256), blk, 0, stream>>>(qbuf, kbuf, csum);

    // 4) L2 normalize over DK (both tensors)
    l2norm_kernel<<<dim3((Bc * Lc * Hc) / 8, 2), blk, 0, stream>>>(qbuf, kbuf);

    // 5) per-head low-rank scores -> directly into d_out slices
    lowrank_kernel<<<dim3(Lc / 64, Hc, Bc), blk, 0, stream>>>(qbuf, We, esc);
    lowrank_kernel<<<dim3(Lc / 64, Hc, Bc), blk, 0, stream>>>(kbuf, Wr, rsc);

    // 6) score @ Wc + bc -> bf16 permuted activation
    mix_kernel<<<dim3(Lc / 64, Bc), blk, 0, stream>>>(esc, rsc, Wc, bc, hh);

    // 7) final projection (WMMA bf16) -> d_out
    gemm_bf16_wmma<<<ggrid, blk, 0, stream>>>(hh, WoT, bo, out);

    // 8) Lam passthrough
    hipMemcpyAsync(lam, Lam, (size_t)Hc * Rc * sizeof(float),
                   hipMemcpyDeviceToDevice, stream);
}
